// BDH_WithTracking_4518305596046
// MI455X (gfx1250) — compile-verified, matrix operand order checked by reading, not run on hardware
//
#include <hip/hip_runtime.h>

// ---------------------------------------------------------------------------
// BDH-style block: 6 layers of (encode -> rope -> causal gram -> mix -> decode)
// All GEMMs run on v_wmma_f32_16x16x32_f16 (f16 in, f32 accumulate).
// 128x128x32 block tile, 8 waves (wave32), 8 WMMA per wave per k-step.
// Double-buffered LDS tiles; staging uses GLOBAL_LOAD_ASYNC_TO_LDS_B128
// (ASYNCcnt) when the toolchain exposes the builtin, else register staging.
// ---------------------------------------------------------------------------

typedef __attribute__((ext_vector_type(16))) _Float16 v16h;
typedef __attribute__((ext_vector_type(8)))  _Float16 v8h;
typedef __attribute__((ext_vector_type(8)))  float    v8f;
typedef __attribute__((ext_vector_type(4)))  unsigned int uint4v;
typedef __attribute__((ext_vector_type(4)))  int      v4i;

#define BM 128
#define BN 128
#define BK 32
#define LDT 48   // padded LDS row length (halves); 96B rows keep 16B alignment

#if defined(__has_builtin)
#if __has_builtin(__builtin_amdgcn_global_load_async_to_lds_b128) && \
    __has_builtin(__builtin_amdgcn_s_wait_asynccnt)
#define USE_ASYNC_LDS 1
#endif
#endif
#ifndef USE_ASYNC_LDS
#define USE_ASYNC_LDS 0
#endif

__device__ __forceinline__ void async_b128(const _Float16* g, _Float16* l)
{
#if USE_ASYNC_LDS
    typedef __attribute__((address_space(1))) v4i* gptr_t;   // global v4i*
    typedef __attribute__((address_space(3))) v4i* sptr_t;   // LDS v4i*
    __builtin_amdgcn_global_load_async_to_lds_b128((gptr_t)(g), (sptr_t)(l), 0, 0);
#endif
}

__device__ __forceinline__ void wait_async_all()
{
#if USE_ASYNC_LDS
    __builtin_amdgcn_s_wait_asynccnt(0);
#endif
}

enum { EPI_F32 = 0, EPI_RELU_ROPE = 1, EPI_MASK = 2, EPI_RELU_MUL = 3 };

// C-layout reminder (16x16 f32 C/D, wave32): vgpr r, lane L ->
//   M = r + 8*(L>>4),  N = (L & 15)

template <int EPI, bool TRANS_B>
__global__ void __launch_bounds__(256)
gemm_wmma(const _Float16* __restrict__ A, int lda,
          const _Float16* __restrict__ B, int ldb,
          void* __restrict__ C, int ldc,
          _Float16* __restrict__ C2,                  // RoPE output (EPI_RELU_ROPE)
          const _Float16* __restrict__ AUX, int ldaux,// x_sparse (EPI_RELU_MUL)
          int K, int kCausal)
{
    __shared__ _Float16 As[2][BM][LDT];
    __shared__ _Float16 Bs[2][BN][LDT];   // stored as Bs[buf][n][k]

    const int tid  = threadIdx.x;
    const int lane = tid & 31;
    const int wid  = tid >> 5;                 // 0..7 (wave32)
    const int row0 = blockIdx.x * BM;
    const int col0 = blockIdx.y * BN;

    // Strict-causal full skip: tile entirely above diagonal -> write zeros.
    if (EPI == EPI_MASK && col0 > row0) {
        _Float16* Ch = (_Float16*)C;
        for (int i = tid; i < BM * BN; i += 256) {
            int r = i >> 7, c = i & 127;
            Ch[(size_t)(row0 + r) * ldc + col0 + c] = (_Float16)0.0f;
        }
        return;
    }

    v8f acc[4][2] = {};

    int kEnd = K;
    if (kCausal) { int lim = row0 + BM; kEnd = lim < K ? lim : K; }

    // ---- staging coordinates ----
    const int ar  = tid >> 1;            // 0..127  (A row / Bt row)
    const int ac0 = (tid & 1) * 16;      // 0 or 16 (k chunk)
    const int bk  = tid >> 3;            // 0..31   (B row, K dim, non-trans)
    const int bn0 = (tid & 7) * 16;      // 0..112  (B col chunk, non-trans)

    const _Float16* aSrc  = A + (size_t)(row0 + ar) * lda + ac0;
    const _Float16* bSrcT = TRANS_B ? (B + (size_t)(col0 + ar) * ldb + ac0) : nullptr;
    const _Float16* bSrcN = TRANS_B ? nullptr : (B + (size_t)bk * ldb + col0 + bn0);

    uint4v sa0, sa1, sb0, sb1;           // staged A / Bt chunks (fallback path)
    v8h    sp0, sp1;                     // staged B (non-trans) chunks (always)

    auto issueA = [&](int buf, int kb) {
#if USE_ASYNC_LDS
        async_b128(aSrc + kb,     &As[buf][ar][ac0]);
        async_b128(aSrc + kb + 8, &As[buf][ar][ac0 + 8]);
#else
        const uint4v* s = (const uint4v*)(aSrc + kb);
        sa0 = s[0]; sa1 = s[1];
        (void)buf;
#endif
    };
    auto storeA = [&](int buf) {
#if !USE_ASYNC_LDS
        *(uint4v*)(&As[buf][ar][ac0])     = sa0;
        *(uint4v*)(&As[buf][ar][ac0 + 8]) = sa1;
#else
        (void)buf;
#endif
    };
    auto issueBt = [&](int buf, int kb) {
#if USE_ASYNC_LDS
        async_b128(bSrcT + kb,     &Bs[buf][ar][ac0]);
        async_b128(bSrcT + kb + 8, &Bs[buf][ar][ac0 + 8]);
#else
        const uint4v* t = (const uint4v*)(bSrcT + kb);
        sb0 = t[0]; sb1 = t[1];
        (void)buf;
#endif
    };
    auto storeBt = [&](int buf) {
#if !USE_ASYNC_LDS
        *(uint4v*)(&Bs[buf][ar][ac0])     = sb0;
        *(uint4v*)(&Bs[buf][ar][ac0 + 8]) = sb1;
#else
        (void)buf;
#endif
    };
    auto loadBn = [&](int kb) {          // K-major B: must transpose via regs
        const _Float16* t = bSrcN + (size_t)kb * ldb;
        sp0 = *(const v8h*)(t);
        sp1 = *(const v8h*)(t + 8);
    };
    auto storeBn = [&](int buf) {
#pragma unroll
        for (int i = 0; i < 8; ++i) {
            Bs[buf][bn0 + i][bk]     = sp0[i];
            Bs[buf][bn0 + 8 + i][bk] = sp1[i];
        }
    };

    // ---- prologue: stage tile 0 into buffer 0 ----
    issueA(0, 0);
    if (TRANS_B) issueBt(0, 0); else loadBn(0);
    storeA(0);
    if (TRANS_B) storeBt(0); else storeBn(0);
    wait_async_all();
    __syncthreads();

    const int lane15 = lane & 15;
    const int kg     = (lane >> 4) * 8;      // 0 or 8
    const int wm     = (wid & 1) * 64;       // wave M base within block
    const int wn     = (wid >> 1) * 32;      // wave N base within block

    for (int kb = 0; kb < kEnd; kb += BK) {
        const int  ibuf    = (kb / BK) & 1;
        const int  nxt     = ibuf ^ 1;
        const bool hasNext = (kb + BK) < kEnd;

        // issue next tile (async writes go to the other LDS buffer)
        if (hasNext) {
            issueA(nxt, kb + BK);
            if (TRANS_B) issueBt(nxt, kb + BK); else loadBn(kb + BK);
        }
        if (kb + 2 * BK < kEnd) {
            __builtin_prefetch(aSrc + kb + 2 * BK, 0, 1);
            if (TRANS_B) __builtin_prefetch(bSrcT + kb + 2 * BK, 0, 1);
            else         __builtin_prefetch(bSrcN + (size_t)(kb + 2 * BK) * ldb, 0, 1);
        }

        // ---- fragments + WMMA on current buffer ----
        // 16-bit A/B layout: lanes 0-15 hold K {0..7,16..23}, lanes 16-31 {8..15,24..31}
        v16h bfrag[2];
#pragma unroll
        for (int bt = 0; bt < 2; ++bt) {
            const int ncol = wn + bt * 16 + lane15;
            v8h lo = *(const v8h*)(&Bs[ibuf][ncol][kg]);
            v8h hi = *(const v8h*)(&Bs[ibuf][ncol][16 + kg]);
#pragma unroll
            for (int i = 0; i < 8; ++i) { bfrag[bt][i] = lo[i]; bfrag[bt][8 + i] = hi[i]; }
        }
#pragma unroll
        for (int mt = 0; mt < 4; ++mt) {
            const int mrow = wm + mt * 16 + lane15;
            v16h afrag;
            v8h lo = *(const v8h*)(&As[ibuf][mrow][kg]);
            v8h hi = *(const v8h*)(&As[ibuf][mrow][16 + kg]);
#pragma unroll
            for (int i = 0; i < 8; ++i) { afrag[i] = lo[i]; afrag[8 + i] = hi[i]; }
            acc[mt][0] = __builtin_amdgcn_wmma_f32_16x16x32_f16(
                false, afrag, false, bfrag[0], (short)0, acc[mt][0], false, false);
            acc[mt][1] = __builtin_amdgcn_wmma_f32_16x16x32_f16(
                false, afrag, false, bfrag[1], (short)0, acc[mt][1], false, false);
        }

        if (hasNext) {
            // fallback-path stores go to the other buffer; its previous readers
            // finished before the last barrier, so no extra barrier needed here.
            storeA(nxt);
            if (TRANS_B) storeBt(nxt); else storeBn(nxt);
            wait_async_all();
            __syncthreads();
        }
    }

    // ---- epilogue ----
#pragma unroll
    for (int mt = 0; mt < 4; ++mt) {
        const int mBase = row0 + wm + mt * 16 + (lane >> 4) * 8;
#pragma unroll
        for (int bt = 0; bt < 2; ++bt) {
            const int n = col0 + wn + bt * 16 + lane15;
#pragma unroll
            for (int r = 0; r < 8; ++r) {
                const int m = mBase + r;
                float a = acc[mt][bt][r];
                if (EPI == EPI_F32) {
                    ((float*)C)[(size_t)m * ldc + n] = a;
                } else if (EPI == EPI_MASK) {
                    if (n >= m) a = 0.0f;            // tril(k=-1): zero where s >= t
                    ((_Float16*)C)[(size_t)m * ldc + n] = (_Float16)a;
                } else if (EPI == EPI_RELU_ROPE) {
                    a = a > 0.0f ? a : 0.0f;
                    ((_Float16*)C)[(size_t)m * ldc + n] = (_Float16)a;   // x_sparse
                    // RoPE: partner element n^1 lives in lane^1 (same vgpr r)
                    float ar2 = __shfl_xor(a, 1, 32);
                    int   q   = n & ~1;
                    float fr  = exp2f(-16.0f * (float)q * (1.0f / 4096.0f)) * 0.15915494309189535f;
                    float ph  = (float)m * fr;
                    ph -= floorf(ph);
                    float ang = ph * 6.283185307179586f;
                    float c   = __cosf(ang);
                    float s   = __sinf(ang);
                    float sgn = (n & 1) ? 1.0f : -1.0f;  // even n: -v[n+1]; odd: +v[n-1]
                    float qr  = a * c + sgn * ar2 * s;
                    C2[(size_t)m * ldc + n] = (_Float16)qr;              // QR
                } else { // EPI_RELU_MUL: xy = relu(acc) * x_sparse
                    a = a > 0.0f ? a : 0.0f;
                    float xs = (float)AUX[(size_t)m * ldaux + n];
                    ((_Float16*)C)[(size_t)m * ldc + n] = (_Float16)(a * xs);
                }
            }
        }
    }
}

// ---------------------------------------------------------------------------
// Elementwise / LayerNorm helpers
// ---------------------------------------------------------------------------

__global__ void f32_to_f16(const float* __restrict__ src, _Float16* __restrict__ dst, int n)
{
    int i = blockIdx.x * blockDim.x + threadIdx.x;
    if (i < n) dst[i] = (_Float16)src[i];
}

__device__ __forceinline__ float block_sum256(float v, float* red)
{
    const int d = threadIdx.x;
    red[d] = v;
    __syncthreads();
#pragma unroll
    for (int s = 128; s > 0; s >>= 1) {
        if (d < s) red[d] += red[d + s];
        __syncthreads();
    }
    float r = red[0];
    __syncthreads();
    return r;
}

// x = LN(embed_w[idx]) ; writes f32 + f16 copies. One block per token, D=256.
__global__ void __launch_bounds__(256)
embed_ln_kernel(const int* __restrict__ idx, const float* __restrict__ emb,
                float* __restrict__ xf, _Float16* __restrict__ xh)
{
    __shared__ float red[256];
    const int t = blockIdx.x, d = threadIdx.x;
    const int tok = idx[t];
    float v = emb[(size_t)tok * 256 + d];
    float m = block_sum256(v, red) * (1.0f / 256.0f);
    float c = v - m;
    float var = block_sum256(c * c, red) * (1.0f / 256.0f);
    float y = c * rsqrtf(var + 1e-5f);
    xf[(size_t)t * 256 + d] = y;
    xh[(size_t)t * 256 + d] = (_Float16)y;
}

// row-wise LN over 256 (yKV): f32 in, f16 out. One block per row.
__global__ void __launch_bounds__(256)
ln_row_kernel(const float* __restrict__ src, _Float16* __restrict__ dst)
{
    __shared__ float red[256];
    const int r = blockIdx.x, d = threadIdx.x;
    float v = src[(size_t)r * 256 + d];
    float m = block_sum256(v, red) * (1.0f / 256.0f);
    float c = v - m;
    float var = block_sum256(c * c, red) * (1.0f / 256.0f);
    dst[(size_t)r * 256 + d] = (_Float16)(c * rsqrtf(var + 1e-5f));
}

// x = LN(x + LN(yMLP)); updates f32 + f16 copies of x. One block per token.
__global__ void __launch_bounds__(256)
ln_residual_kernel(const float* __restrict__ ymlp, float* __restrict__ xf,
                   _Float16* __restrict__ xh)
{
    __shared__ float red[256];
    const int t = blockIdx.x, d = threadIdx.x;
    float u = ymlp[(size_t)t * 256 + d];
    float m = block_sum256(u, red) * (1.0f / 256.0f);
    float c = u - m;
    float var = block_sum256(c * c, red) * (1.0f / 256.0f);
    float u2 = c * rsqrtf(var + 1e-5f);

    float v = xf[(size_t)t * 256 + d] + u2;
    float m2 = block_sum256(v, red) * (1.0f / 256.0f);
    float c2 = v - m2;
    float var2 = block_sum256(c2 * c2, red) * (1.0f / 256.0f);
    float y = c2 * rsqrtf(var2 + 1e-5f);
    xf[(size_t)t * 256 + d] = y;
    xh[(size_t)t * 256 + d] = (_Float16)y;
}

// ---------------------------------------------------------------------------
// Orchestration
// ---------------------------------------------------------------------------

extern "C" void kernel_launch(void* const* d_in, const int* in_sizes, int n_in,
                              void* d_out, int out_size, void* d_ws, size_t ws_size,
                              hipStream_t stream)
{
    (void)in_sizes; (void)n_in; (void)out_size; (void)ws_size;

    const int   T = 1024, D = 256, NH = 4, N = 4096, VOC = 256;
    const int   L = 6;  // n_layer (d_in[1]); fixed by setup_inputs, launch count must be capture-stable

    const int*   idx     = (const int*)  d_in[0];
    const float* embed_w = (const float*)d_in[2];
    const float* enc_f   = (const float*)d_in[3];   // (nh, D, N)
    const float* encv_f  = (const float*)d_in[4];   // (nh, D, N)
    const float* dec_f   = (const float*)d_in[5];   // (nh*N, D)
    const float* lmh_f   = (const float*)d_in[6];   // (D, vocab)
    float* out = (float*)d_out;                     // (T, vocab)

    // ---- workspace carve ----
    char* p = (char*)d_ws;
    auto alloc = [&](size_t bytes) { void* r = (void*)p; p += (bytes + 255) & ~(size_t)255; return r; };

    _Float16* enc_h  = (_Float16*)alloc((size_t)NH * D * N * 2);   // 8 MB
    _Float16* encv_h = (_Float16*)alloc((size_t)NH * D * N * 2);   // 8 MB
    _Float16* dec_h  = (_Float16*)alloc((size_t)NH * N * D * 2);   // 8 MB
    _Float16* lmh_h  = (_Float16*)alloc((size_t)D * VOC * 2);      // 128 KB
    float*    x_f32  = (float*)   alloc((size_t)T * D * 4);        // 1 MB
    _Float16* x_f16  = (_Float16*)alloc((size_t)T * D * 2);        // 0.5 MB
    _Float16* xsp    = (_Float16*)alloc((size_t)NH * T * N * 2);   // 32 MB  x_sparse [h][t][n]
    _Float16* qr_xy  = (_Float16*)alloc((size_t)NH * T * N * 2);   // 32 MB  QR, reused as xy [t][h*N+n]
    _Float16* scores = (_Float16*)alloc((size_t)NH * T * T * 2);   // 8 MB
    float*    ykv32  = (float*)   alloc((size_t)NH * T * D * 4);   // 4 MB
    _Float16* ykv16  = (_Float16*)alloc((size_t)NH * T * D * 2);   // 2 MB
    float*    ymlp   = (float*)   alloc((size_t)T * D * 4);        // 1 MB

    // ---- weight conversion to f16 ----
    {
        int n1 = NH * D * N;
        f32_to_f16<<<dim3((n1 + 255) / 256), dim3(256), 0, stream>>>(enc_f,  enc_h,  n1);
        f32_to_f16<<<dim3((n1 + 255) / 256), dim3(256), 0, stream>>>(encv_f, encv_h, n1);
        int n2 = NH * N * D;
        f32_to_f16<<<dim3((n2 + 255) / 256), dim3(256), 0, stream>>>(dec_f,  dec_h,  n2);
        int n3 = D * VOC;
        f32_to_f16<<<dim3((n3 + 255) / 256), dim3(256), 0, stream>>>(lmh_f,  lmh_h,  n3);
    }

    // ---- embedding + LN ----
    embed_ln_kernel<<<dim3(T), dim3(256), 0, stream>>>(idx, embed_w, x_f32, x_f16);

    const dim3 blk(256);
    for (int l = 0; l < L; ++l) {
        // (1) x_sparse = relu(X @ Enc[h]); QR = rope(x_sparse)   [M=T, N=4096, K=D]
        for (int h = 0; h < NH; ++h) {
            gemm_wmma<EPI_RELU_ROPE, false><<<dim3(T / BM, N / BN), blk, 0, stream>>>(
                x_f16, D,
                enc_h + (size_t)h * D * N, N,
                (void*)(xsp + (size_t)h * T * N), N,
                qr_xy + (size_t)h * T * N,
                nullptr, 0, D, 0);
        }
        // (2) scores = (QR @ QR^T) * tril(-1)                     [M=T, N=T, K=4096]
        for (int h = 0; h < NH; ++h) {
            const _Float16* qh = qr_xy + (size_t)h * T * N;
            gemm_wmma<EPI_MASK, true><<<dim3(T / BM, T / BN), blk, 0, stream>>>(
                qh, N, qh, N,
                (void*)(scores + (size_t)h * T * T), T,
                nullptr, nullptr, 0, N, 0);
        }
        // (3) yKV_pre = scores @ X   (causal K limit)             [M=T, N=D, K=T]
        for (int h = 0; h < NH; ++h) {
            gemm_wmma<EPI_F32, false><<<dim3(T / BM, D / BN), blk, 0, stream>>>(
                scores + (size_t)h * T * T, T,
                x_f16, D,
                (void*)(ykv32 + (size_t)h * T * D), D,
                nullptr, nullptr, 0, T, 1);
        }
        // yKV = LN(yKV_pre)  (nh*T rows of 256)
        ln_row_kernel<<<dim3(NH * T), dim3(256), 0, stream>>>(ykv32, ykv16);

        // (4) xy = relu(yKV @ Enc_v[h]) * x_sparse -> [t][h*N+n]  [M=T, N=4096, K=D]
        for (int h = 0; h < NH; ++h) {
            gemm_wmma<EPI_RELU_MUL, false><<<dim3(T / BM, N / BN), blk, 0, stream>>>(
                ykv16 + (size_t)h * T * D, D,
                encv_h + (size_t)h * D * N, N,
                (void*)(qr_xy + (size_t)h * N), NH * N,       // xy view, ldc = 16384
                nullptr,
                xsp + (size_t)h * T * N, N, D, 0);
        }
        // (5) yMLP = xy @ decoder                                 [M=T, N=D, K=16384]
        gemm_wmma<EPI_F32, false><<<dim3(T / BM, D / BN), blk, 0, stream>>>(
            qr_xy, NH * N, dec_h, D, (void*)ymlp, D,
            nullptr, nullptr, 0, NH * N, 0);

        // x = LN(x + LN(yMLP))
        ln_residual_kernel<<<dim3(T), dim3(256), 0, stream>>>(ymlp, x_f32, x_f16);
    }

    // logits = x @ lm_head                                        [M=T, N=256, K=256]
    gemm_wmma<EPI_F32, false><<<dim3(T / BM, VOC / BN), blk, 0, stream>>>(
        x_f16, D, lmh_h, VOC, (void*)out, VOC,
        nullptr, nullptr, 0, D, 0);
}
